// LocalitySelfAttention_17403207483576
// MI455X (gfx1250) — matrix operand order repaired and checked by reading, not compiled, not run on hardware
//
#include <hip/hip_runtime.h>

// MI455X (gfx1250, wave32) bf16-WMMA implementation of causal multi-head attention.
// Pipeline: cvt x->bf16, transpose weights, QKV GEMM -> Q/K/Vt, flash attention, out GEMM.

typedef __attribute__((ext_vector_type(16))) __bf16 v16bf;
typedef __attribute__((ext_vector_type(8)))  __bf16 v8bf;
typedef __attribute__((ext_vector_type(4)))  __bf16 v4bf;
typedef __attribute__((ext_vector_type(8)))  float  v8f;

#define B_ 8
#define N_ 1024
#define C_ 1024
#define H_ 16
#define D_ 64

__device__ __forceinline__ v16bf cat8(v8bf lo, v8bf hi) {
  return __builtin_shufflevector(lo, hi, 0,1,2,3,4,5,6,7,8,9,10,11,12,13,14,15);
}

// A-matrix 16x32 bf16 fragment (row-major source [M][K], stride in elements).
// lane<16: M=lane, elems 0..15 = K 0..15
// lane>=16: M=lane-16, elems 0..15 = K 16..31
__device__ __forceinline__ v16bf load_a_frag(const __bf16* base, int stride, int m0, int k0, int lane) {
  int hp = (lane >> 4) & 1, l = lane & 15;
  const __bf16* p = base + (size_t)(m0 + l) * stride + (k0 + hp * 16);
  v8bf lo = *(const v8bf*)(p);
  v8bf hi = *(const v8bf*)(p + 8);
  return cat8(lo, hi);
}

// B-matrix 32x16 bf16 fragment from K-contiguous source [N][K].
// lane<16: N=lane, K 0..15; lane>=16: N=lane-16, K 16..31
__device__ __forceinline__ v16bf load_b_frag(const __bf16* base, int stride, int n0, int k0, int lane) {
  int hp = (lane >> 4) & 1, l = lane & 15;
  const __bf16* p = base + (size_t)(n0 + l) * stride + (k0 + hp * 16);
  v8bf lo = *(const v8bf*)(p);
  v8bf hi = *(const v8bf*)(p + 8);
  return cat8(lo, hi);
}

__device__ __forceinline__ v8f wmma_bf16(v16bf a, v16bf b, v8f c) {
  // (neg_a, A, neg_b, B, c_mod, C, reuse_a, reuse_b)
  return __builtin_amdgcn_wmma_f32_16x16x32_bf16(false, a, false, b, (short)0, c, false, false);
}

// ---------------- elementwise convert ----------------
__global__ __launch_bounds__(256) void cvt_f32_bf16(const float* __restrict__ in,
                                                    __bf16* __restrict__ out, int n) {
  int i = (blockIdx.x * blockDim.x + threadIdx.x) * 4;
  if (i >= n) return;
  float4 f = *(const float4*)(in + i);
  v4bf o = { (__bf16)f.x, (__bf16)f.y, (__bf16)f.z, (__bf16)f.w };
  *(v4bf*)(out + i) = o;
}

// ---------------- tiled transpose + convert: in[R][Cc] f32 -> outT[Cc][R] bf16 ----------------
__global__ __launch_bounds__(256) void transpose_cvt(const float* __restrict__ in,
                                                     __bf16* __restrict__ outT, int R, int Cc) {
  __shared__ float t[32][33];
  int c0 = blockIdx.x * 32, r0 = blockIdx.y * 32;
  int tx = threadIdx.x & 31, ty = threadIdx.x >> 5;
  for (int i = ty; i < 32; i += 8) t[i][tx] = in[(size_t)(r0 + i) * Cc + (c0 + tx)];
  __syncthreads();
  for (int i = ty; i < 32; i += 8) outT[(size_t)(c0 + i) * R + (r0 + tx)] = (__bf16)t[tx][i];
}

// ---------------- QKV GEMM: [8192,1024] x [1024,3072] -> Q/K [B,H,N,D], Vt [B,H,D,N] ----------------
__global__ __launch_bounds__(256) void gemm_qkv_kernel(const __bf16* __restrict__ xb,
                                                       const __bf16* __restrict__ wT,
                                                       __bf16* __restrict__ Q,
                                                       __bf16* __restrict__ K,
                                                       __bf16* __restrict__ Vt) {
  int lane = threadIdx.x & 31, wid = threadIdx.x >> 5;
  int wm = wid & 3, wn = wid >> 2;
  int m0 = blockIdx.y * 128 + wm * 32;
  int n0 = blockIdx.x * 128 + wn * 64;
  v8f acc[2][4];
#pragma unroll
  for (int i = 0; i < 2; i++)
#pragma unroll
    for (int j = 0; j < 4; j++) acc[i][j] = {};

  for (int k = 0; k < C_; k += 32) {
    if (k + 32 < C_) {
      __builtin_prefetch((const void*)(xb + (size_t)m0 * C_ + k + 32), 0, 1);
      __builtin_prefetch((const void*)(wT + (size_t)n0 * C_ + k + 32), 0, 1);
    }
    v16bf a0 = load_a_frag(xb, C_, m0,      k, lane);
    v16bf a1 = load_a_frag(xb, C_, m0 + 16, k, lane);
    v16bf b0 = load_b_frag(wT, C_, n0,      k, lane);
    v16bf b1 = load_b_frag(wT, C_, n0 + 16, k, lane);
    v16bf b2 = load_b_frag(wT, C_, n0 + 32, k, lane);
    v16bf b3 = load_b_frag(wT, C_, n0 + 48, k, lane);
    acc[0][0] = wmma_bf16(a0, b0, acc[0][0]);
    acc[0][1] = wmma_bf16(a0, b1, acc[0][1]);
    acc[0][2] = wmma_bf16(a0, b2, acc[0][2]);
    acc[0][3] = wmma_bf16(a0, b3, acc[0][3]);
    acc[1][0] = wmma_bf16(a1, b0, acc[1][0]);
    acc[1][1] = wmma_bf16(a1, b1, acc[1][1]);
    acc[1][2] = wmma_bf16(a1, b2, acc[1][2]);
    acc[1][3] = wmma_bf16(a1, b3, acc[1][3]);
  }

  int hp = lane >> 4, li = lane & 15;
#pragma unroll
  for (int i = 0; i < 2; i++) {
#pragma unroll
    for (int j = 0; j < 4; j++) {
      int n = n0 + j * 16 + li;
      int sec = n >> 10;     // 0=Q, 1=K, 2=V
      int c = n & 1023;
      int h = c >> 6, d = c & 63;
#pragma unroll
      for (int v = 0; v < 8; v++) {
        int m = m0 + i * 16 + v + hp * 8;
        int bb = m >> 10, row = m & 1023;
        __bf16 val = (__bf16)acc[i][j][v];
        if (sec == 0)      Q[(size_t)((bb * H_ + h) * N_ + row) * D_ + d] = val;
        else if (sec == 1) K[(size_t)((bb * H_ + h) * N_ + row) * D_ + d] = val;
        else               Vt[(size_t)((bb * H_ + h) * D_ + d) * N_ + row] = val;
      }
    }
  }
}

// ---------------- flash attention, strict causal (j < i), per-head scale ----------------
__global__ __launch_bounds__(256) void attn_kernel(const __bf16* __restrict__ Q,
                                                   const __bf16* __restrict__ K,
                                                   const __bf16* __restrict__ Vt,
                                                   const float* __restrict__ scale,
                                                   __bf16* __restrict__ ctx) {
  __shared__ __bf16 Pl[8][16 * 32];   // per-wave P staging (C-layout -> A-layout re-fragment)
  int lane = threadIdx.x & 31, wid = threadIdx.x >> 5;
  int hp = lane >> 4, li = lane & 15;
  int bh = blockIdx.y;
  int b = bh >> 4, h = bh & 15;
  int q0 = blockIdx.x * 128 + wid * 16;
  const __bf16* Qh = Q  + (size_t)bh * N_ * D_;
  const __bf16* Kh = K  + (size_t)bh * N_ * D_;
  const __bf16* Vh = Vt + (size_t)bh * D_ * N_;
  float hscale = scale[h];

  v16bf qa0 = load_a_frag(Qh, D_, q0, 0,  lane);
  v16bf qa1 = load_a_frag(Qh, D_, q0, 32, lane);

  v8f acc0 = {}, acc1 = {}, acc2 = {}, acc3 = {};
  float l[8], rmax[8];
#pragma unroll
  for (int v = 0; v < 8; v++) { l[v] = 0.f; rmax[v] = -3.0e38f; }

  int ntile = (q0 + 14) / 32 + 1;     // tiles of 32 keys, last needed key = q0+14
  __bf16* Pw = &Pl[wid][0];

  for (int t = 0; t < ntile; t++) {
    int kj = t * 32;
    v16bf kb00 = load_b_frag(Kh, D_, kj,      0,  lane);
    v16bf kb01 = load_b_frag(Kh, D_, kj,      32, lane);
    v16bf kb10 = load_b_frag(Kh, D_, kj + 16, 0,  lane);
    v16bf kb11 = load_b_frag(Kh, D_, kj + 16, 32, lane);
    v8f s0 = {}, s1 = {};
    s0 = wmma_bf16(qa0, kb00, s0);
    s0 = wmma_bf16(qa1, kb01, s0);
    s1 = wmma_bf16(qa0, kb10, s1);
    s1 = wmma_bf16(qa1, kb11, s1);

    int n0k = kj + li, n1k = kj + 16 + li;
#pragma unroll
    for (int v = 0; v < 8; v++) {
      int m = q0 + v + hp * 8;
      bool a0 = n0k < m, a1 = n1k < m;
      float v0 = a0 ? s0[v] * hscale : -3.0e38f;
      float v1 = a1 ? s1[v] * hscale : -3.0e38f;
      // row max across the 16-lane half-wave holding this row
      float mx = fmaxf(v0, v1);
      mx = fmaxf(mx, __shfl_xor(mx, 1, 32));
      mx = fmaxf(mx, __shfl_xor(mx, 2, 32));
      mx = fmaxf(mx, __shfl_xor(mx, 4, 32));
      mx = fmaxf(mx, __shfl_xor(mx, 8, 32));
      float nm = fmaxf(rmax[v], mx);
      float corr = __expf(rmax[v] - nm);
      rmax[v] = nm;
      float e0 = a0 ? __expf(v0 - nm) : 0.f;
      float e1 = a1 ? __expf(v1 - nm) : 0.f;
      float rs = e0 + e1;
      rs += __shfl_xor(rs, 1, 32);
      rs += __shfl_xor(rs, 2, 32);
      rs += __shfl_xor(rs, 4, 32);
      rs += __shfl_xor(rs, 8, 32);
      l[v] = l[v] * corr + rs;
      acc0[v] *= corr; acc1[v] *= corr; acc2[v] *= corr; acc3[v] *= corr;
      int prow = v + hp * 8;
      Pw[prow * 32 + li]      = (__bf16)e0;
      Pw[prow * 32 + 16 + li] = (__bf16)e1;
    }
    // P (16x32) as A-fragment from LDS; V as B-fragments from Vt [D][N]
    v16bf pa  = load_a_frag(Pw, 32, 0, 0, lane);
    v16bf vb0 = load_b_frag(Vh, N_, 0,  kj, lane);
    v16bf vb1 = load_b_frag(Vh, N_, 16, kj, lane);
    v16bf vb2 = load_b_frag(Vh, N_, 32, kj, lane);
    v16bf vb3 = load_b_frag(Vh, N_, 48, kj, lane);
    acc0 = wmma_bf16(pa, vb0, acc0);
    acc1 = wmma_bf16(pa, vb1, acc1);
    acc2 = wmma_bf16(pa, vb2, acc2);
    acc3 = wmma_bf16(pa, vb3, acc3);
  }

#pragma unroll
  for (int v = 0; v < 8; v++) {
    float inv = (l[v] > 0.f) ? 1.f / l[v] : 0.f;   // fully-masked row -> 0 (flash convention)
    int m = q0 + v + hp * 8;
    size_t rowbase = ((size_t)b * N_ + m) * C_ + h * D_;
    ctx[rowbase +  0 + li] = (__bf16)(acc0[v] * inv);
    ctx[rowbase + 16 + li] = (__bf16)(acc1[v] * inv);
    ctx[rowbase + 32 + li] = (__bf16)(acc2[v] * inv);
    ctx[rowbase + 48 + li] = (__bf16)(acc3[v] * inv);
  }
}

// ---------------- out GEMM: ctx[8192,1024] x w_outT -> f32 + bias ----------------
__global__ __launch_bounds__(256) void gemm_out_kernel(const __bf16* __restrict__ A,
                                                       const __bf16* __restrict__ wT,
                                                       const float* __restrict__ bias,
                                                       float* __restrict__ out) {
  int lane = threadIdx.x & 31, wid = threadIdx.x >> 5;
  int wm = wid & 3, wn = wid >> 2;
  int m0 = blockIdx.y * 128 + wm * 32;
  int n0 = blockIdx.x * 128 + wn * 64;
  v8f acc[2][4];
#pragma unroll
  for (int i = 0; i < 2; i++)
#pragma unroll
    for (int j = 0; j < 4; j++) acc[i][j] = {};

  for (int k = 0; k < C_; k += 32) {
    if (k + 32 < C_) {
      __builtin_prefetch((const void*)(A  + (size_t)m0 * C_ + k + 32), 0, 1);
      __builtin_prefetch((const void*)(wT + (size_t)n0 * C_ + k + 32), 0, 1);
    }
    v16bf a0 = load_a_frag(A, C_, m0,      k, lane);
    v16bf a1 = load_a_frag(A, C_, m0 + 16, k, lane);
    v16bf b0 = load_b_frag(wT, C_, n0,      k, lane);
    v16bf b1 = load_b_frag(wT, C_, n0 + 16, k, lane);
    v16bf b2 = load_b_frag(wT, C_, n0 + 32, k, lane);
    v16bf b3 = load_b_frag(wT, C_, n0 + 48, k, lane);
    acc[0][0] = wmma_bf16(a0, b0, acc[0][0]);
    acc[0][1] = wmma_bf16(a0, b1, acc[0][1]);
    acc[0][2] = wmma_bf16(a0, b2, acc[0][2]);
    acc[0][3] = wmma_bf16(a0, b3, acc[0][3]);
    acc[1][0] = wmma_bf16(a1, b0, acc[1][0]);
    acc[1][1] = wmma_bf16(a1, b1, acc[1][1]);
    acc[1][2] = wmma_bf16(a1, b2, acc[1][2]);
    acc[1][3] = wmma_bf16(a1, b3, acc[1][3]);
  }

  int hp = lane >> 4, li = lane & 15;
#pragma unroll
  for (int i = 0; i < 2; i++) {
#pragma unroll
    for (int j = 0; j < 4; j++) {
      int n = n0 + j * 16 + li;
      float bn = bias[n];
#pragma unroll
      for (int v = 0; v < 8; v++) {
        int m = m0 + i * 16 + v + hp * 8;
        out[(size_t)m * C_ + n] = acc[i][j][v] + bn;
      }
    }
  }
}

extern "C" void kernel_launch(void* const* d_in, const int* in_sizes, int n_in,
                              void* d_out, int out_size, void* d_ws, size_t ws_size,
                              hipStream_t stream) {
  (void)in_sizes; (void)n_in; (void)out_size; (void)ws_size;
  const float* x      = (const float*)d_in[0];  // [8,1024,1024]
  const float* w_qkv  = (const float*)d_in[1];  // [1024,3072]
  const float* scale  = (const float*)d_in[2];  // [16,1,1]
  const float* w_out  = (const float*)d_in[3];  // [1024,1024]
  const float* b_out  = (const float*)d_in[4];  // [1024]
  float* out = (float*)d_out;

  char* ws = (char*)d_ws;
  const size_t MB = 1024 * 1024;
  __bf16* xb    = (__bf16*)(ws + 0);        // 16MB, reused as ctx after QKV GEMM consumes it
  __bf16* ctx   = xb;
  __bf16* wqkvT = (__bf16*)(ws + 16 * MB);  // 6MB  [3072][1024]
  __bf16* woutT = (__bf16*)(ws + 22 * MB);  // 2MB  [1024][1024]
  __bf16* Q     = (__bf16*)(ws + 24 * MB);  // 16MB [B,H,N,D]
  __bf16* K     = (__bf16*)(ws + 40 * MB);  // 16MB [B,H,N,D]
  __bf16* Vt    = (__bf16*)(ws + 56 * MB);  // 16MB [B,H,D,N]

  cvt_f32_bf16<<<dim3(8192), dim3(256), 0, stream>>>(x, xb, B_ * N_ * C_);
  transpose_cvt<<<dim3(96, 32), dim3(256), 0, stream>>>(w_qkv, wqkvT, C_, 3 * C_);
  transpose_cvt<<<dim3(32, 32), dim3(256), 0, stream>>>(w_out, woutT, C_, C_);
  gemm_qkv_kernel<<<dim3(24, 64), dim3(256), 0, stream>>>(xb, wqkvT, Q, K, Vt);
  attn_kernel<<<dim3(8, 128), dim3(256), 0, stream>>>(Q, K, Vt, scale, ctx);
  gemm_out_kernel<<<dim3(8, 64), dim3(256), 0, stream>>>(ctx, woutT, b_out, out);
}
